// MHAttention_35278861369472
// MI455X (gfx1250) — matrix operand order repaired and checked by reading, not compiled, not run on hardware
//
#include <hip/hip_runtime.h>
#include <hip/hip_bf16.h>

// ---------------------------------------------------------------------------
// Problem constants (match reference)
// ---------------------------------------------------------------------------
#define BATCH 2
#define SEQ   2048
#define DIM   1024
#define HEADS 16
#define DH    64
#define MASK_VALUE (-1e-06f)

// ---------------------------------------------------------------------------
// WMMA types (CDNA5 / gfx1250, wave32)
// ---------------------------------------------------------------------------
typedef __attribute__((ext_vector_type(16))) __bf16 v16bf;
typedef __attribute__((ext_vector_type(8)))  float  v8f;

union Frag16 {              // 16 bf16 values = 8 VGPRs
    v16bf    v;
    unsigned u[8];
};

__device__ __forceinline__ v8f wmma_bf16(const Frag16& a, const Frag16& b, v8f c) {
    // D = A(16x32 bf16) * B(32x16 bf16) + C(16x16 f32)
    return __builtin_amdgcn_wmma_f32_16x16x32_bf16(
        /*neg_a=*/false, a.v, /*neg_b=*/false, b.v,
        /*c_mod=*/(short)0, c, /*reuse_a=*/false, /*reuse_b=*/false);
}

// fp32 -> bf16 with round-to-nearest-even (bit pattern as ushort)
__device__ __forceinline__ unsigned short f2bf(float f) {
    unsigned u = __float_as_uint(f);
    unsigned r = u + 0x7FFFu + ((u >> 16) & 1u);
    return (unsigned short)(r >> 16);
}

// ---------------------------------------------------------------------------
// gfx1250 async global->LDS copy (128b per lane, ASYNCcnt-tracked, no VGPR
// data round-trip). LDS operand is the low 32 bits of the flat address
// ---------------------------------------------------------------------------
__device__ __forceinline__ void g2lds_b128(const void* g, void* l) {
    unsigned loff = (unsigned)(uintptr_t)l;
    asm volatile("global_load_async_to_lds_b128 %0, %1, off"
                 :: "v"(loff), "v"(g) : "memory");
}
__device__ __forceinline__ void wait_async0() {
    asm volatile("s_wait_asynccnt 0x0" ::: "memory");
}

// ---------------------------------------------------------------------------
// Kernel: fp32 -> bf16 bulk convert (8 elements / thread)
// ---------------------------------------------------------------------------
__global__ void cvt_f32_bf16(const float* __restrict__ src,
                             unsigned short* __restrict__ dst, int n) {
    int i = (blockIdx.x * blockDim.x + threadIdx.x) * 8;
    if (i >= n) return;
    float4 a = ((const float4*)(src + i))[0];
    float4 b = ((const float4*)(src + i))[1];
    uint4 o;
    o.x = (unsigned)f2bf(a.x) | ((unsigned)f2bf(a.y) << 16);
    o.y = (unsigned)f2bf(a.z) | ((unsigned)f2bf(a.w) << 16);
    o.z = (unsigned)f2bf(b.x) | ((unsigned)f2bf(b.y) << 16);
    o.w = (unsigned)f2bf(b.z) | ((unsigned)f2bf(b.w) << 16);
    *(uint4*)(dst + i) = o;
}

// fp32 (K x N) -> bf16 TRANSPOSED (N x K): one-time weight prep so GEMM B
// tiles become natural-layout async copies.
__global__ void cvt_t_f32_bf16(const float* __restrict__ src,
                               unsigned short* __restrict__ dst) {
    int idx = blockIdx.x * blockDim.x + threadIdx.x;  // 0 .. DIM*DIM-1
    int r = idx >> 10, c = idx & 1023;
    dst[(size_t)c * DIM + r] = f2bf(src[idx]);
}

// ---------------------------------------------------------------------------
// Fold bilinear W (64x64, shared over heads) into Wq; output TRANSPOSED bf16.
// WqW[k][64h + j] = sum_i Wq[k][64h + i] * W[i][j]
// ---------------------------------------------------------------------------
__global__ void fold_wq(const float* __restrict__ Wq, const float* __restrict__ W,
                        unsigned short* __restrict__ outT) {
    int idx = blockIdx.x * blockDim.x + threadIdx.x;  // 0 .. DIM*DIM-1
    int row = idx >> 10;          // input dim k
    int c   = idx & 1023;         // output col
    int hh = c >> 6, cj = c & 63;
    const float* wqrow = Wq + (size_t)row * DIM + hh * DH;
    float s = 0.f;
#pragma unroll 8
    for (int i = 0; i < DH; ++i) s = fmaf(wqrow[i], W[i * DH + cj], s);
    outT[(size_t)c * DIM + row] = f2bf(s);
}

__global__ void fold_bq(const float* __restrict__ bq, const float* __restrict__ W,
                        float* __restrict__ out) {
    int c = blockIdx.x * blockDim.x + threadIdx.x;
    if (c >= DIM) return;
    int hh = c >> 6, cj = c & 63;
    float s = 0.f;
    for (int i = 0; i < DH; ++i) s = fmaf(bq[hh * DH + i], W[i * DH + cj], s);
    out[c] = s;
}

// ---------------------------------------------------------------------------
// Tiled bf16 GEMM with WMMA: C[M,N] = A[M,K] @ Bt^T[K,N] + bias
// (Bt is pre-transposed N x K.)  Block tile 128x128, K-step 32, 256 threads =
// 8 waves (2x4), each wave 64x32 via 4x2 grid of 16x16 WMMA tiles.
// Double-buffered LDS, async global->LDS staging, one barrier per K-step.
// store_mode: 0 = bf16 row-major, 1 = f32 row-major,
//             2 = bf16 per-head transposed V layout (B,H,DH,S)
// ---------------------------------------------------------------------------
#define BM 128
#define BN 128
#define BK 32

__global__ __launch_bounds__(256)
void gemm_bf16(const unsigned short* __restrict__ A,
               const unsigned short* __restrict__ Bt,
               const float* __restrict__ bias,
               float* __restrict__ Cf, unsigned short* __restrict__ Cb,
               int M, int N, int K, int store_mode) {
    __shared__ unsigned short As[2][BM][BK];
    __shared__ unsigned short Bs[2][BN][BK];  // [n][k]

    const int tid  = threadIdx.x;
    const int lane = tid & 31;
    const int wave = tid >> 5;
    const int wm = wave >> 2;          // 0..1  (64 rows each)
    const int wn = wave & 3;           // 0..3  (32 cols each)
    const int l15 = lane & 15, lh = lane >> 4;
    const int bm0 = blockIdx.y * BM, bn0 = blockIdx.x * BN;

    const int srow = tid >> 1, sseg = (tid & 1) * 16;

    auto stage = [&](int buf, int k0) {
        const unsigned short* as = A  + (size_t)(bm0 + srow) * K + k0 + sseg;
        const unsigned short* bs = Bt + (size_t)(bn0 + srow) * K + k0 + sseg;
        g2lds_b128(as,     &As[buf][srow][sseg]);
        g2lds_b128(as + 8, &As[buf][srow][sseg + 8]);
        g2lds_b128(bs,     &Bs[buf][srow][sseg]);
        g2lds_b128(bs + 8, &Bs[buf][srow][sseg + 8]);
    };

    v8f acc[4][2] = {};

    stage(0, 0);
    int buf = 0;
    for (int k0 = 0; k0 < K; k0 += BK) {
        wait_async0();
        __syncthreads();
        if (k0 + BK < K) stage(buf ^ 1, k0 + BK);

        Frag16 af[4], bf[2];
#pragma unroll
        for (int i = 0; i < 4; ++i) {
            int m = wm * 64 + i * 16 + l15;
            int kb = lh * 8;
#pragma unroll
            for (int r = 0; r < 8; ++r) {
                int kk = kb + (r & 3) * 2 + (r >> 2) * 16;
                af[i].u[r] = *(const unsigned*)&As[buf][m][kk];
            }
        }
#pragma unroll
        for (int j = 0; j < 2; ++j) {
            int n = wn * 32 + j * 16 + l15;
            int kb = lh * 8;
#pragma unroll
            for (int r = 0; r < 8; ++r)
                bf[j].u[r] = *(const unsigned*)&Bs[buf][n][kb + (r & 3) * 2 + (r >> 2) * 16];
        }
#pragma unroll
        for (int i = 0; i < 4; ++i)
#pragma unroll
            for (int j = 0; j < 2; ++j)
                acc[i][j] = wmma_bf16(af[i], bf[j], acc[i][j]);
        buf ^= 1;
    }

    // ---- epilogue: C/D layout lane n = lane&15, row = r + 8*(lane>>4) ----
#pragma unroll
    for (int i = 0; i < 4; ++i) {
#pragma unroll
        for (int j = 0; j < 2; ++j) {
            int n = bn0 + wn * 32 + j * 16 + l15;
            float bv = bias ? bias[n] : 0.f;
#pragma unroll
            for (int r = 0; r < 8; ++r) {
                int m = bm0 + wm * 64 + i * 16 + r + lh * 8;
                float v = acc[i][j][r] + bv;
                if (store_mode == 1) {
                    Cf[(size_t)m * N + n] = v;
                } else if (store_mode == 0) {
                    Cb[(size_t)m * N + n] = f2bf(v);
                } else {  // V projection written per-head transposed (B,H,DH,S)
                    int bb = m >> 11, ss = m & (SEQ - 1);
                    int hh = n >> 6,  dd = n & 63;
                    Cb[(((size_t)bb * HEADS + hh) * DH + dd) * SEQ + ss] = f2bf(v);
                }
            }
        }
    }
}

// ---------------------------------------------------------------------------
// Flash-style attention: one block per (b, h, 64-row q tile); 4 waves, each
// wave owns 16 q rows. Double-buffered K/V tiles staged with async copies
// (V comes pre-transposed per head, so both tiles are natural-layout copies).
// S = Qw @ K^T (bf16 WMMA, f32 acc), value-replace mask (-1e-6), online
// softmax via 16-lane xor-shuffle reductions, P LDS round-trip to A-layout,
// O += P @ V. Output bf16, merged-head (B,S,D).
// ---------------------------------------------------------------------------
__global__ __launch_bounds__(128)
void attn_kernel(const unsigned short* __restrict__ Qw,
                 const unsigned short* __restrict__ Kp,
                 const unsigned short* __restrict__ Vt,   // (B,H,DH,S)
                 const int* __restrict__ thr,
                 unsigned short* __restrict__ Obuf) {
    __shared__ unsigned short Ks[2][64][DH];    // [kpos][dh]
    __shared__ unsigned short Vs[2][DH][64];    // [dh][kpos]
    __shared__ unsigned short Ps[4][16][64];    // per-wave P tile

    const int qt = blockIdx.x;                  // 0..S/64-1
    const int bh = blockIdx.y;                  // 0..B*H-1
    const int b = bh >> 4, h = bh & 15;
    const int tid = threadIdx.x, lane = tid & 31, wave = tid >> 5;
    const int l15 = lane & 15, lh = lane >> 4;
    const int q0 = qt * 64 + wave * 16;
    const int thresh = thr[b];

    const unsigned short* Qbase = Qw + ((size_t)b * SEQ) * DIM + (size_t)h * DH;
    const unsigned short* Kbase = Kp + ((size_t)b * SEQ) * DIM + (size_t)h * DH;
    const unsigned short* Vtb   = Vt + (size_t)bh * DH * SEQ;

    const int srow = tid >> 1, sseg = (tid & 1) * 32;

    auto stage = [&](int bufi, int kt) {
        const unsigned short* ksrc = Kbase + (size_t)(kt * 64 + srow) * DIM + sseg;
        const unsigned short* vsrc = Vtb + (size_t)srow * SEQ + kt * 64 + sseg;
#pragma unroll
        for (int t = 0; t < 32; t += 8) {
            g2lds_b128(ksrc + t, &Ks[bufi][srow][sseg + t]);
            g2lds_b128(vsrc + t, &Vs[bufi][srow][sseg + t]);
        }
    };

    // Q A-fragments (16 rows x 64 dh -> two 16x32 frags), loaded once
    Frag16 aq[2];
    {
        const unsigned short* qrow = Qbase + (size_t)(q0 + l15) * DIM;
        int kb = lh * 8;
#pragma unroll
        for (int c = 0; c < 2; ++c)
#pragma unroll
            for (int r = 0; r < 8; ++r) {
                int kk = c * 32 + kb + (r & 3) * 2 + (r >> 2) * 16;
                aq[c].u[r] = *(const unsigned*)(qrow + kk);
            }
    }

    float rm[8], rl[8];
    v8f accO[4] = {};
#pragma unroll
    for (int r = 0; r < 8; ++r) { rm[r] = -3.0e38f; rl[r] = 0.f; }

    stage(0, 0);
    int buf = 0;
    for (int kt = 0; kt < SEQ / 64; ++kt) {
        wait_async0();
        __syncthreads();
        if (kt + 1 < SEQ / 64) stage(buf ^ 1, kt + 1);

        // ---- scores: 16x64 tile = Qfrag @ Ktile^T ----
        v8f sc[4];
#pragma unroll
        for (int j = 0; j < 4; ++j) {
            Frag16 b0, b1;
            int n = j * 16 + l15;                // key position within tile
#pragma unroll
            for (int r = 0; r < 8; ++r) {
                int kk = lh * 8 + (r & 3) * 2 + (r >> 2) * 16;        // dh index
                b0.u[r] = *(const unsigned*)&Ks[buf][n][kk];
                b1.u[r] = *(const unsigned*)&Ks[buf][n][32 + kk];
            }
            v8f c = {};
            c = wmma_bf16(aq[0], b0, c);
            c = wmma_bf16(aq[1], b1, c);
            sc[j] = c;
        }

        // ---- mask + online softmax (rows r + 8*lh; cols across 16 lanes) ----
#pragma unroll
        for (int r = 0; r < 8; ++r) {
            float mx = -3.0e38f;
#pragma unroll
            for (int j = 0; j < 4; ++j) {
                int kp = kt * 64 + j * 16 + l15;
                float v = sc[j][r];
                v = (kp < thresh) ? v : MASK_VALUE;   // value-replace, per reference
                sc[j][r] = v;
                mx = fmaxf(mx, v);
            }
#pragma unroll
            for (int d = 1; d < 16; d <<= 1)
                mx = fmaxf(mx, __shfl_xor(mx, d, 32));
            float nm = fmaxf(rm[r], mx);
            float scale = __expf(rm[r] - nm);
            rm[r] = nm;
            float s = 0.f;
#pragma unroll
            for (int j = 0; j < 4; ++j) {
                float p = __expf(sc[j][r] - nm);
                sc[j][r] = p;
                s += p;
            }
#pragma unroll
            for (int d = 1; d < 16; d <<= 1)
                s += __shfl_xor(s, d, 32);
            rl[r] = rl[r] * scale + s;
#pragma unroll
            for (int j = 0; j < 4; ++j)
                accO[j][r] *= scale;
        }

        // ---- P: C-layout -> LDS -> A-layout (per-wave region, wave-local) ----
#pragma unroll
        for (int j = 0; j < 4; ++j)
#pragma unroll
            for (int r = 0; r < 8; ++r)
                Ps[wave][r + lh * 8][j * 16 + l15] = f2bf(sc[j][r]);
        asm volatile("s_wait_dscnt 0x0" ::: "memory");

        Frag16 ap[2];
        {
            int kb = lh * 8;
#pragma unroll
            for (int c = 0; c < 2; ++c)
#pragma unroll
                for (int r = 0; r < 8; ++r) {
                    int kk = c * 32 + kb + (r & 3) * 2 + (r >> 2) * 16;
                    ap[c].u[r] = *(const unsigned*)&Ps[wave][l15][kk];
                }
        }

        // ---- O += P @ V ----
#pragma unroll
        for (int j = 0; j < 4; ++j) {
            Frag16 bv0, bv1;
            int n = j * 16 + l15;                // dh index
#pragma unroll
            for (int r = 0; r < 8; ++r) {
                int kk = lh * 8 + (r & 3) * 2 + (r >> 2) * 16;        // key position
                bv0.u[r] = *(const unsigned*)&Vs[buf][n][kk];
                bv1.u[r] = *(const unsigned*)&Vs[buf][n][32 + kk];
            }
            accO[j] = wmma_bf16(ap[0], bv0, accO[j]);
            accO[j] = wmma_bf16(ap[1], bv1, accO[j]);
        }
        buf ^= 1;
    }

    // ---- normalize + store bf16 merged-head output ----
    unsigned short* Ob = Obuf + ((size_t)b * SEQ) * DIM + (size_t)h * DH;
#pragma unroll
    for (int j = 0; j < 4; ++j)
#pragma unroll
        for (int r = 0; r < 8; ++r) {
            int m = q0 + r + lh * 8;
            float v = accO[j][r] / rl[r];
            Ob[(size_t)m * DIM + j * 16 + l15] = f2bf(v);
        }
}

// ---------------------------------------------------------------------------
// Host launcher
// ---------------------------------------------------------------------------
extern "C" void kernel_launch(void* const* d_in, const int* in_sizes, int n_in,
                              void* d_out, int out_size, void* d_ws, size_t ws_size,
                              hipStream_t stream) {
    const float* queries = (const float*)d_in[0];
    const float* keys    = (const float*)d_in[1];
    const float* values  = (const float*)d_in[2];
    const int*   thrs    = (const int*)  d_in[3];
    const float* Wq = (const float*)d_in[4];
    const float* bq = (const float*)d_in[5];
    const float* Wk = (const float*)d_in[6];
    const float* bk = (const float*)d_in[7];
    const float* Wv = (const float*)d_in[8];
    const float* bv = (const float*)d_in[9];
    const float* W  = (const float*)d_in[10];
    const float* Wo = (const float*)d_in[11];
    const float* bo = (const float*)d_in[12];
    float* out = (float*)d_out;

    const size_t ACT = (size_t)BATCH * SEQ * DIM;   // 4,194,304
    const size_t WGT = (size_t)DIM * DIM;           // 1,048,576

    char* w = (char*)d_ws;
    auto carve = [&](size_t bytes) {
        char* p = w;
        w += (bytes + 255) & ~(size_t)255;
        return p;
    };
    unsigned short* qb     = (unsigned short*)carve(ACT * 2);
    unsigned short* kb     = (unsigned short*)carve(ACT * 2);
    unsigned short* vbuf   = (unsigned short*)carve(ACT * 2);
    unsigned short* wqT    = (unsigned short*)carve(WGT * 2);  // transposed (N x K)
    unsigned short* wkT    = (unsigned short*)carve(WGT * 2);
    unsigned short* wvT    = (unsigned short*)carve(WGT * 2);
    unsigned short* woT    = (unsigned short*)carve(WGT * 2);
    float*          bqW    = (float*)         carve(DIM * 4);
    unsigned short* Qw     = (unsigned short*)carve(ACT * 2);
    unsigned short* Kpj    = (unsigned short*)carve(ACT * 2);
    unsigned short* Vt     = (unsigned short*)carve(ACT * 2);  // (B,H,DH,S)
    unsigned short* Obuf   = (unsigned short*)carve(ACT * 2);

    // 1) convert activations to bf16; weights to bf16 TRANSPOSED; fold W into Wq
    cvt_f32_bf16<<<ACT / (8 * 256), 256, 0, stream>>>(queries, qb,   (int)ACT);
    cvt_f32_bf16<<<ACT / (8 * 256), 256, 0, stream>>>(keys,    kb,   (int)ACT);
    cvt_f32_bf16<<<ACT / (8 * 256), 256, 0, stream>>>(values,  vbuf, (int)ACT);
    cvt_t_f32_bf16<<<WGT / 256, 256, 0, stream>>>(Wk, wkT);
    cvt_t_f32_bf16<<<WGT / 256, 256, 0, stream>>>(Wv, wvT);
    cvt_t_f32_bf16<<<WGT / 256, 256, 0, stream>>>(Wo, woT);
    fold_wq<<<WGT / 256, 256, 0, stream>>>(Wq, W, wqT);
    fold_bq<<<DIM / 256, 256, 0, stream>>>(bq, W, bqW);

    // 2) projections: Qw = q @ (Wq@blkdiag(W)) + bq@W ; K = k@Wk + bk ;
    //    V = v@Wv + bv written per-head transposed
    dim3 ggrid(DIM / BN, (BATCH * SEQ) / BM);
    gemm_bf16<<<ggrid, 256, 0, stream>>>(qb,   wqT, bqW, nullptr, Qw,  BATCH * SEQ, DIM, DIM, 0);
    gemm_bf16<<<ggrid, 256, 0, stream>>>(kb,   wkT, bk,  nullptr, Kpj, BATCH * SEQ, DIM, DIM, 0);
    gemm_bf16<<<ggrid, 256, 0, stream>>>(vbuf, wvT, bv,  nullptr, Vt,  BATCH * SEQ, DIM, DIM, 2);

    // 3) attention (flash-style, online softmax, threshold mask)
    dim3 agrid(SEQ / 64, BATCH * HEADS);
    attn_kernel<<<agrid, 128, 0, stream>>>(Qw, Kpj, Vt, thrs, Obuf);

    // 4) output projection: out = O @ Wo + bo (fp32 out)
    gemm_bf16<<<ggrid, 256, 0, stream>>>(Obuf, woT, bo, out, nullptr, BATCH * SEQ, DIM, DIM, 1);
}
